// ViT_34282428956707
// MI455X (gfx1250) — matrix operand order repaired and checked by reading, not compiled, not run on hardware
//
#include <hip/hip_runtime.h>
#include <hip/hip_bf16.h>

typedef __attribute__((ext_vector_type(16))) _Float16 v16h;
typedef __attribute__((ext_vector_type(8)))  _Float16 v8h;
typedef __attribute__((ext_vector_type(8)))  float    v8f;
typedef _Float16 f16;

#define DD     384
#define HH     6
#define HDIM   64
#define DEPTH  12
#define NTOK   197
#define NTP    224           // tokens padded to multiple of 32
#define BB     64
#define DFF    1536
#define NCLS   1000
#define NCP    1024          // head cols padded to multiple of 64
#define G14    14
#define NPATCH 196
#define PKDIM  768           // 3*16*16
#define ROWS   (BB*NTP)      // 14336
#define PROWS  (BB*NPATCH)   // 12544
#define ASCALE 0.125f        // 64^-0.5
#define LNEPS  1e-5f

// ---------------- WMMA fragment loads (CDNA5 layouts, see CDNA5 ISA) --------
// A (16x32 f16, row-major src): lane -> M = row+(lane&15);
//   halves: K = kb..kb+7 and kb+16..kb+23, kb = (lane&16)?8:0  -> two 16B loads
__device__ __forceinline__ v16h loadA(const f16* A, int lda, int row, int k0) {
  int lane = (int)threadIdx.x;
  const f16* p = A + (size_t)(row + (lane & 15)) * lda + (size_t)(k0 + ((lane & 16) ? 8 : 0));
  union { v16h v; v8h h[2]; } u;
  u.h[0] = *(const v8h*)(p);
  u.h[1] = *(const v8h*)(p + 16);
  return u.v;
}
// B (32x16 f16, row-major src KxN): lane -> N = col+(lane&15);
//   K = kb..kb+7 and kb+16..kb+23, kb = (lane&16)?8:0 (matches loadA)
__device__ __forceinline__ v16h loadB(const f16* B, int ldb, int col, int k0) {
  int lane = (int)threadIdx.x;
  const f16* p = B + (size_t)(k0 + ((lane & 16) ? 8 : 0)) * ldb + col + (lane & 15);
  v16h v;
#pragma unroll
  for (int s = 0; s < 8; ++s) {
    v[s] = p[(size_t)s * ldb];
    v[s + 8] = p[(size_t)(s + 16) * ldb];
  }
  return v;
}

#define WMMA_F16(a,b,c) __builtin_amdgcn_wmma_f32_16x16x32_f16(false,(a),false,(b),(short)0,(c),false,false)

// 32x32 output tile per wave (attention GEMMs, M/N multiple of 32).
__device__ __forceinline__ void gemm32(const f16* A, int lda, const f16* B, int ldb,
                                       int K, int row0, int col0, v8f c[2][2]) {
  v8f z = {0.f,0.f,0.f,0.f,0.f,0.f,0.f,0.f};
  c[0][0]=z; c[0][1]=z; c[1][0]=z; c[1][1]=z;
  for (int k0 = 0; k0 < K; k0 += 32) {
    v16h a0 = loadA(A, lda, row0,      k0);
    v16h a1 = loadA(A, lda, row0 + 16, k0);
    v16h b0 = loadB(B, ldb, col0,      k0);
    v16h b1 = loadB(B, ldb, col0 + 16, k0);
    c[0][0] = WMMA_F16(a0, b0, c[0][0]);
    c[0][1] = WMMA_F16(a0, b1, c[0][1]);
    c[1][0] = WMMA_F16(a1, b0, c[1][0]);
    c[1][1] = WMMA_F16(a1, b1, c[1][1]);
  }
}

// 32x64 output tile per wave: 8 accumulators (64 VGPRs), 4 b128 A-loads + 64 16-bit B-loads per
// 8 wmma per K-step (0.75 KB/wmma). Safe register pressure (~120 VGPRs),
// unlike the 64x64 variant which spilled accumulators to scratch.
__device__ __forceinline__ void gemm3264(const f16* A, int lda, const f16* B, int ldb,
                                         int K, int row0, int col0, v8f c[2][4]) {
  v8f z = {0.f,0.f,0.f,0.f,0.f,0.f,0.f,0.f};
#pragma unroll
  for (int i = 0; i < 2; ++i)
#pragma unroll
    for (int j = 0; j < 4; ++j) c[i][j] = z;
  for (int k0 = 0; k0 < K; k0 += 32) {
    v16h a0 = loadA(A, lda, row0,      k0);
    v16h a1 = loadA(A, lda, row0 + 16, k0);
#pragma unroll
    for (int j = 0; j < 4; ++j) {
      v16h b = loadB(B, ldb, col0 + 16 * j, k0);
      c[0][j] = WMMA_F16(a0, b, c[0][j]);
      c[1][j] = WMMA_F16(a1, b, c[1][j]);
    }
  }
}

// D layout: lane -> N = col0+16j+(lane&15); VGPR r -> M = row0+16i+r+((lane&16)?8:0)
#define EPILOGUE2(c, ...) do {                                             \
  int lane_ = (int)threadIdx.x;                                            \
  int mAdd_ = (lane_ & 16) ? 8 : 0;  int nLow_ = lane_ & 15;               \
  for (int i_ = 0; i_ < 2; ++i_) for (int j_ = 0; j_ < 2; ++j_)            \
    for (int r_ = 0; r_ < 8; ++r_) {                                       \
      int m = row0 + 16*i_ + r_ + mAdd_;                                   \
      int n = col0 + 16*j_ + nLow_;                                        \
      float val = c[i_][j_][r_];                                           \
      { __VA_ARGS__; }                                                     \
    }                                                                      \
} while (0)

#define EPILOGUE24(c, ...) do {                                            \
  int lane_ = (int)threadIdx.x;                                            \
  int mAdd_ = (lane_ & 16) ? 8 : 0;  int nLow_ = lane_ & 15;               \
  for (int i_ = 0; i_ < 2; ++i_) for (int j_ = 0; j_ < 4; ++j_)            \
    for (int r_ = 0; r_ < 8; ++r_) {                                       \
      int m = row0 + 16*i_ + r_ + mAdd_;                                   \
      int n = col0 + 16*j_ + nLow_;                                        \
      float val = c[i_][j_][r_];                                           \
      { __VA_ARGS__; }                                                     \
    }                                                                      \
} while (0)

// ---------------- elementwise / prep kernels --------------------------------
__global__ void k_convert(const float* in, f16* out, long long n) {
  long long i = (long long)blockIdx.x * blockDim.x + threadIdx.x;
  if (i < n) out[i] = (f16)in[i];
}
__global__ void k_convert_head(const float* in, f16* out) {  // 384x1000 -> 384x1024 (pad stays 0)
  int i = blockIdx.x * blockDim.x + threadIdx.x;
  if (i < DD * NCLS) { int r = i / NCLS, c = i % NCLS; out[r * NCP + c] = (f16)in[i]; }
}
__global__ void k_patch(const float* x, f16* P) {  // (B*196, 768) f16 patch matrix
  long long i = (long long)blockIdx.x * blockDim.x + threadIdx.x;
  if (i >= (long long)PROWS * PKDIM) return;
  int col = (int)(i % PKDIM); long long row = i / PKDIM;
  int b = (int)(row / NPATCH), p = (int)(row % NPATCH);
  int c = col >> 8, rem = col & 255, py = rem >> 4, px = rem & 15;
  int gy = p / G14, gx = p % G14;
  long long xi = (((long long)b * 3 + c) * 224 + (gy * 16 + py)) * 224 + (gx * 16 + px);
  P[i] = (f16)x[xi];
}

// ---------------- GEMM kernels (row-space: 32x64 tiles) ----------------------
__global__ void k_gemm_patch(const f16* P, const f16* W, const float* bias, float* pe) {
  int row0 = 32 * (blockIdx.y * 4 + threadIdx.y), col0 = 64 * blockIdx.x;
  if (row0 >= PROWS) return;
  v8f c[2][4]; gemm3264(P, PKDIM, W, DD, PKDIM, row0, col0, c);
  EPILOGUE24(c, pe[(size_t)m * DD + n] = val + bias[n]);
}

__global__ void k_gemm_q(const f16* a16, const f16* Wq, f16* q16) {
  int row0 = 32 * (blockIdx.y * 4 + threadIdx.y), col0 = 64 * blockIdx.x;
  if (row0 >= ROWS) return;
  v8f c[2][4]; gemm3264(a16, DD, Wq, DD, DD, row0, col0, c);
  EPILOGUE24(c, {
    int b = m / NTP, t = m % NTP, hh = n >> 6, dd = n & 63;
    q16[(((size_t)b * HH + hh) * NTP + t) * HDIM + dd] = (f16)val;
  });
}

__global__ void k_gemm_kv(const f16* a16, const f16* Wkv, f16* kt16, f16* v16) {
  int row0 = 32 * (blockIdx.y * 4 + threadIdx.y), col0 = 64 * blockIdx.x;
  if (row0 >= ROWS) return;
  v8f c[2][4]; gemm3264(a16, DD, Wkv, 2 * DD, DD, row0, col0, c);
  EPILOGUE24(c, {
    int b = m / NTP, t = m % NTP;
    if (n < DD) { int hh = n >> 6, dd = n & 63;                 // K stored d-major (transposed)
      kt16[(((size_t)b * HH + hh) * HDIM + dd) * NTP + t] = (f16)val;
    } else { int n2 = n - DD; int hh = n2 >> 6, dd = n2 & 63;   // V stored (t,d)
      v16[(((size_t)b * HH + hh) * NTP + t) * HDIM + dd] = (f16)val;
    }
  });
}

__global__ void k_gemm_o(const f16* o16, const f16* Wo, const float* bo, float* h) {
  int row0 = 32 * (blockIdx.y * 4 + threadIdx.y), col0 = 64 * blockIdx.x;
  if (row0 >= ROWS) return;
  v8f c[2][4]; gemm3264(o16, DD, Wo, DD, DD, row0, col0, c);
  EPILOGUE24(c, if ((m % NTP) < NTOK) h[(size_t)m * DD + n] += val + bo[n]);
}

__global__ void k_gemm_mlp1(const f16* a16, const f16* W1, const float* b1, f16* mid) {
  int row0 = 32 * (blockIdx.y * 4 + threadIdx.y), col0 = 64 * blockIdx.x;
  if (row0 >= ROWS) return;
  v8f c[2][4]; gemm3264(a16, DD, W1, DFF, DD, row0, col0, c);
  EPILOGUE24(c, {
    float xg = val + b1[n];
    float g = 0.5f * xg * (1.f + erff(xg * 0.70710678118654752f));  // exact GELU
    mid[(size_t)m * DFF + n] = ((m % NTP) < NTOK) ? (f16)g : (f16)0.f;
  });
}

__global__ void k_gemm_mlp2(const f16* mid, const f16* W2, const float* b2, float* h) {
  int row0 = 32 * (blockIdx.y * 4 + threadIdx.y), col0 = 64 * blockIdx.x;
  if (row0 >= ROWS) return;
  v8f c[2][4]; gemm3264(mid, DFF, W2, DD, DFF, row0, col0, c);
  EPILOGUE24(c, if ((m % NTP) < NTOK) h[(size_t)m * DD + n] += val + b2[n]);
}

__global__ void k_gemm_head(const f16* clsa, const f16* Wh, const float* bh, float* out) {
  int row0 = 32 * (blockIdx.y * 4 + threadIdx.y), col0 = 64 * blockIdx.x;
  if (row0 >= BB) return;   // M=64: waves 0,1 active
  v8f c[2][4]; gemm3264(clsa, DD, Wh, NCP, DD, row0, col0, c);
  EPILOGUE24(c, if (n < NCLS) out[(size_t)m * NCLS + n] = val + bh[n]);
}

// ---------------- attention GEMMs (32x32 tiles, M=N=224) ---------------------
__global__ void k_gemm_scores(const f16* q16, const f16* kt16, f16* sim) {
  int z = blockIdx.z;
  int row0 = 32 * (blockIdx.y * 4 + threadIdx.y), col0 = 32 * blockIdx.x;
  if (row0 >= NTP) return;
  const f16* A = q16  + (size_t)z * NTP * HDIM;
  const f16* B = kt16 + (size_t)z * HDIM * NTP;
  v8f c[2][2]; gemm32(A, HDIM, B, NTP, HDIM, row0, col0, c);
  f16* S = sim + (size_t)z * NTP * NTP;
  EPILOGUE2(c, S[(size_t)m * NTP + n] = (f16)(val * ASCALE));
}

__global__ void k_softmax(f16* sim) {
  __shared__ float sm[256];
  int z = blockIdx.x, t = blockIdx.y, tid = threadIdx.x;
  f16* row = sim + ((size_t)z * NTP + t) * NTP;
  if (t >= NTOK) { if (tid < NTP) row[tid] = (f16)0.f; return; }
  float s = (tid < NTOK) ? (float)row[tid] : -1e30f;
  sm[tid] = s; __syncthreads();
  for (int k = 128; k > 0; k >>= 1) { if (tid < k) sm[tid] = fmaxf(sm[tid], sm[tid + k]); __syncthreads(); }
  float mx = sm[0]; __syncthreads();
  float e = (tid < NTOK) ? expf(s - mx) : 0.f;
  sm[tid] = e; __syncthreads();
  for (int k = 128; k > 0; k >>= 1) { if (tid < k) sm[tid] += sm[tid + k]; __syncthreads(); }
  float inv = 1.f / sm[0];
  if (tid < NTP) row[tid] = (tid < NTOK) ? (f16)(e * inv) : (f16)0.f;
}

__global__ void k_gemm_av(const f16* sim, const f16* v16, f16* o16) {
  int z = blockIdx.z;
  int row0 = 32 * (blockIdx.y * 4 + threadIdx.y), col0 = 32 * blockIdx.x;
  if (row0 >= NTP) return;
  const f16* A = sim + (size_t)z * NTP * NTP;
  const f16* B = v16 + (size_t)z * NTP * HDIM;
  v8f c[2][2]; gemm32(A, NTP, B, HDIM, NTP, row0, col0, c);
  int b = z / HH, hh = z % HH;
  EPILOGUE2(c, o16[((size_t)b * NTP + m) * DD + hh * HDIM + n] = (f16)val);
}

// ---------------- LayerNorm kernels (fp32, 128 threads x 3 features) ---------
__global__ void k_ln0(const float* pe, const float* w, const float* bia,
                      const float* pos, const float* clstok, float* h) {
  __shared__ float sm[128];
  int b = blockIdx.x, t = blockIdx.y, tid = threadIdx.x;
  float* hrow = h + ((size_t)b * NTP + t) * DD;
  if (t == 0) {
    for (int f = tid; f < DD; f += 128) hrow[f] = clstok[f] + pos[f];
    return;
  }
  const float* xr = pe + ((size_t)b * NPATCH + (t - 1)) * DD;
  float x0 = xr[tid], x1 = xr[tid + 128], x2 = xr[tid + 256];
  sm[tid] = x0 + x1 + x2; __syncthreads();
  for (int k = 64; k > 0; k >>= 1) { if (tid < k) sm[tid] += sm[tid + k]; __syncthreads(); }
  float mean = sm[0] * (1.f / DD); __syncthreads();
  float d0 = x0 - mean, d1 = x1 - mean, d2 = x2 - mean;
  sm[tid] = d0 * d0 + d1 * d1 + d2 * d2; __syncthreads();
  for (int k = 64; k > 0; k >>= 1) { if (tid < k) sm[tid] += sm[tid + k]; __syncthreads(); }
  float rs = rsqrtf(sm[0] * (1.f / DD) + LNEPS);
  const float* pr = pos + (size_t)t * DD;
  hrow[tid]       = d0 * rs * w[tid]       + bia[tid]       + pr[tid];
  hrow[tid + 128] = d1 * rs * w[tid + 128] + bia[tid + 128] + pr[tid + 128];
  hrow[tid + 256] = d2 * rs * w[tid + 256] + bia[tid + 256] + pr[tid + 256];
}

__global__ void k_ln(const float* h, const float* w, const float* bia, f16* out) {
  __shared__ float sm[128];
  int b = blockIdx.x, t = blockIdx.y, tid = threadIdx.x;
  f16* orow = out + ((size_t)b * NTP + t) * DD;
  if (t >= NTOK) { orow[tid] = (f16)0.f; orow[tid + 128] = (f16)0.f; orow[tid + 256] = (f16)0.f; return; }
  const float* xr = h + ((size_t)b * NTP + t) * DD;
  float x0 = xr[tid], x1 = xr[tid + 128], x2 = xr[tid + 256];
  sm[tid] = x0 + x1 + x2; __syncthreads();
  for (int k = 64; k > 0; k >>= 1) { if (tid < k) sm[tid] += sm[tid + k]; __syncthreads(); }
  float mean = sm[0] * (1.f / DD); __syncthreads();
  float d0 = x0 - mean, d1 = x1 - mean, d2 = x2 - mean;
  sm[tid] = d0 * d0 + d1 * d1 + d2 * d2; __syncthreads();
  for (int k = 64; k > 0; k >>= 1) { if (tid < k) sm[tid] += sm[tid + k]; __syncthreads(); }
  float rs = rsqrtf(sm[0] * (1.f / DD) + LNEPS);
  orow[tid]       = (f16)(d0 * rs * w[tid]       + bia[tid]);
  orow[tid + 128] = (f16)(d1 * rs * w[tid + 128] + bia[tid + 128]);
  orow[tid + 256] = (f16)(d2 * rs * w[tid + 256] + bia[tid + 256]);
}

__global__ void k_lnf(const float* h, const float* w, const float* bia, f16* outa) {
  __shared__ float sm[128];
  int b = blockIdx.x, tid = threadIdx.x;
  const float* xr = h + (size_t)b * NTP * DD;   // token 0 = CLS
  float x0 = xr[tid], x1 = xr[tid + 128], x2 = xr[tid + 256];
  sm[tid] = x0 + x1 + x2; __syncthreads();
  for (int k = 64; k > 0; k >>= 1) { if (tid < k) sm[tid] += sm[tid + k]; __syncthreads(); }
  float mean = sm[0] * (1.f / DD); __syncthreads();
  float d0 = x0 - mean, d1 = x1 - mean, d2 = x2 - mean;
  sm[tid] = d0 * d0 + d1 * d1 + d2 * d2; __syncthreads();
  for (int k = 64; k > 0; k >>= 1) { if (tid < k) sm[tid] += sm[tid + k]; __syncthreads(); }
  float rs = rsqrtf(sm[0] * (1.f / DD) + LNEPS);
  f16* orow = outa + (size_t)b * DD;
  orow[tid]       = (f16)(d0 * rs * w[tid]       + bia[tid]);
  orow[tid + 128] = (f16)(d1 * rs * w[tid + 128] + bia[tid + 128]);
  orow[tid + 256] = (f16)(d2 * rs * w[tid + 256] + bia[tid + 256]);
}

// ---------------- host launch -------------------------------------------------
extern "C" void kernel_launch(void* const* d_in, const int* in_sizes, int n_in,
                              void* d_out, int out_size, void* d_ws, size_t ws_size,
                              hipStream_t stream) {
  (void)in_sizes; (void)n_in; (void)out_size;
  const float* x    = (const float*)d_in[0];
  const float* Wp   = (const float*)d_in[1];
  const float* bp   = (const float*)d_in[2];
  const float* ln0w = (const float*)d_in[3];
  const float* ln0b = (const float*)d_in[4];
  const float* pos  = (const float*)d_in[5];
  const float* cls  = (const float*)d_in[6];
  const float* ln1w = (const float*)d_in[7];
  const float* ln1b = (const float*)d_in[8];
  const float* Wq   = (const float*)d_in[9];
  const float* Wkv  = (const float*)d_in[10];
  const float* Wo   = (const float*)d_in[11];
  const float* bo   = (const float*)d_in[12];
  const float* ln2w = (const float*)d_in[13];
  const float* ln2b = (const float*)d_in[14];
  const float* W1   = (const float*)d_in[15];
  const float* b1   = (const float*)d_in[16];
  const float* W2   = (const float*)d_in[17];
  const float* b2   = (const float*)d_in[18];
  const float* lnfw = (const float*)d_in[19];
  const float* lnfb = (const float*)d_in[20];
  const float* Wh   = (const float*)d_in[21];
  const float* bh   = (const float*)d_in[22];
  float* out = (float*)d_out;

  char* base = (char*)d_ws; size_t off = 0;
  auto alloc = [&](size_t bytes) -> void* {
    void* p = base + off; off += (bytes + 255) & ~(size_t)255; return p;
  };
  f16*   wp16  = (f16*)alloc((size_t)PKDIM * DD * 2);
  f16*   wq16  = (f16*)alloc((size_t)DEPTH * DD * DD * 2);
  f16*   wkv16 = (f16*)alloc((size_t)DEPTH * DD * 2 * DD * 2);
  f16*   wo16  = (f16*)alloc((size_t)DEPTH * DD * DD * 2);
  f16*   w116  = (f16*)alloc((size_t)DEPTH * DD * DFF * 2);
  f16*   w216  = (f16*)alloc((size_t)DEPTH * DFF * DD * 2);
  f16*   wh16  = (f16*)alloc((size_t)DD * NCP * 2);
  f16*   Pm    = (f16*)alloc((size_t)PROWS * PKDIM * 2);
  float* pe    = (float*)alloc((size_t)PROWS * DD * 4);
  float* h     = (float*)alloc((size_t)ROWS * DD * 4);
  f16*   a16   = (f16*)alloc((size_t)ROWS * DD * 2);
  f16*   q16   = (f16*)alloc((size_t)BB * HH * NTP * HDIM * 2);
  f16*   kt16  = (f16*)alloc((size_t)BB * HH * HDIM * NTP * 2);
  f16*   v16   = (f16*)alloc((size_t)BB * HH * NTP * HDIM * 2);
  f16*   sim   = (f16*)alloc((size_t)BB * HH * NTP * NTP * 2);
  f16*   o16   = (f16*)alloc((size_t)ROWS * DD * 2);
  f16*   mid   = (f16*)alloc((size_t)ROWS * DFF * 2);
  f16*   clsa  = (f16*)alloc((size_t)BB * DD * 2);

  size_t used = off < ws_size ? off : ws_size;
  hipMemsetAsync(d_ws, 0, used, stream);   // zero pads / residual base (capture-safe)

  auto cvt = [&](const float* in, f16* o, size_t n) {
    k_convert<<<dim3((unsigned)((n + 255) / 256)), dim3(256), 0, stream>>>(in, o, (long long)n);
  };
  cvt(Wp,  wp16,  (size_t)PKDIM * DD);
  cvt(Wq,  wq16,  (size_t)DEPTH * DD * DD);
  cvt(Wkv, wkv16, (size_t)DEPTH * DD * 2 * DD);
  cvt(Wo,  wo16,  (size_t)DEPTH * DD * DD);
  cvt(W1,  w116,  (size_t)DEPTH * DD * DFF);
  cvt(W2,  w216,  (size_t)DEPTH * DFF * DD);
  k_convert_head<<<dim3((DD * NCLS + 255) / 256), dim3(256), 0, stream>>>(Wh, wh16);
  k_patch<<<dim3((unsigned)(((size_t)PROWS * PKDIM + 255) / 256)), dim3(256), 0, stream>>>(x, Pm);

  dim3 blk(32, 4);
  k_gemm_patch<<<dim3(DD / 64, PROWS / 128), blk, 0, stream>>>(Pm, wp16, bp, pe);
  k_ln0<<<dim3(BB, NTOK), dim3(128), 0, stream>>>(pe, ln0w, ln0b, pos, cls, h);

  for (int d = 0; d < DEPTH; ++d) {
    k_ln<<<dim3(BB, NTP), dim3(128), 0, stream>>>(h, ln1w + d * DD, ln1b + d * DD, a16);
    k_gemm_q <<<dim3(DD / 64, ROWS / 128), blk, 0, stream>>>(a16, wq16 + (size_t)d * DD * DD, q16);
    k_gemm_kv<<<dim3(2 * DD / 64, ROWS / 128), blk, 0, stream>>>(a16, wkv16 + (size_t)d * DD * 2 * DD, kt16, v16);
    k_gemm_scores<<<dim3(NTP / 32, 2, BB * HH), blk, 0, stream>>>(q16, kt16, sim);
    k_softmax<<<dim3(BB * HH, NTP), dim3(256), 0, stream>>>(sim);
    k_gemm_av<<<dim3(HDIM / 32, 2, BB * HH), blk, 0, stream>>>(sim, v16, o16);
    k_gemm_o<<<dim3(DD / 64, ROWS / 128), blk, 0, stream>>>(o16, wo16 + (size_t)d * DD * DD, bo + d * DD, h);
    k_ln<<<dim3(BB, NTP), dim3(128), 0, stream>>>(h, ln2w + d * DD, ln2b + d * DD, a16);
    k_gemm_mlp1<<<dim3(DFF / 64, ROWS / 128), blk, 0, stream>>>(a16, w116 + (size_t)d * DD * DFF, b1 + d * DFF, mid);
    k_gemm_mlp2<<<dim3(DD / 64, ROWS / 128), blk, 0, stream>>>(mid, w216 + (size_t)d * DFF * DD, b2 + d * DD, h);
  }

  k_lnf<<<dim3(BB), dim3(128), 0, stream>>>(h, lnfw, lnfb, clsa);
  k_gemm_head<<<dim3(NCP / 64, 1), blk, 0, stream>>>(clsa, wh16, bh, out);
}
